// T5Gemma2MergedAttention_6090263626528
// MI455X (gfx1250) — matrix operand order repaired and checked by reading, not compiled, not run on hardware
//
#include <hip/hip_runtime.h>
#include <stdint.h>

typedef __attribute__((ext_vector_type(16))) __bf16 v16bf;
typedef __attribute__((ext_vector_type(8)))  float  v8f;

union Frag { v16bf v; uint4 q[2]; unsigned short s[16]; };

__device__ __forceinline__ unsigned short f2bf(float f) {
  unsigned int x = __float_as_uint(f);
  unsigned int r = (x + 0x7FFFu + ((x >> 16) & 1u)) >> 16;   // RNE
  return (unsigned short)r;
}

// ---------------------------------------------------------------------------
// Straight convert: f32 -> bf16
// ---------------------------------------------------------------------------
__global__ __launch_bounds__(256)
void ccvt_k(const float* __restrict__ X, unsigned short* __restrict__ Y, size_t total) {
  size_t idx = (size_t)blockIdx.x * blockDim.x + threadIdx.x;
  if (idx >= total) return;
  Y[idx] = f2bf(X[idx]);
}

// ---------------------------------------------------------------------------
// Transpose + convert: W[K][N] f32  ->  Wt[N][K] bf16   (write-coalesced)
// ---------------------------------------------------------------------------
__global__ __launch_bounds__(256)
void tcvt_k(const float* __restrict__ W, unsigned short* __restrict__ Wt,
            int K, int N) {
  size_t idx = (size_t)blockIdx.x * blockDim.x + threadIdx.x;
  size_t total = (size_t)K * N;
  if (idx >= total) return;
  unsigned int n = (unsigned int)(idx / (unsigned int)K);
  unsigned int k = (unsigned int)(idx % (unsigned int)K);
  Wt[idx] = f2bf(W[(size_t)k * N + n]);
}

// ---------------------------------------------------------------------------
// GEMM:  C[M,N] = A[M,K] * Bt[N,K]^T   (bf16 WMMA, f32 accumulate)
// A bf16 [M,K]; Bt bf16 [N,K]. C is f32 if OUTF32 else bf16.
// 256 threads = 8 waves; wave tile 32x64; block tile 128x128; K step 32.
// Requires M%128==0, N%128==0, K%32==0.
// ---------------------------------------------------------------------------
template<int OUTF32>
__global__ __launch_bounds__(256)
void gemm_k(const unsigned short* __restrict__ A, const unsigned short* __restrict__ Bt,
            void* __restrict__ Cp, int M, int N, int K) {
  const int lane = threadIdx.x & 31;
  const int wid  = threadIdx.x >> 5;
  const int wm   = wid & 3;          // 4 waves along M
  const int wn   = wid >> 2;         // 2 waves along N
  const int m0   = blockIdx.y * 128 + wm * 32;
  const int n0   = blockIdx.x * 128 + wn * 64;
  const int l15  = lane & 15;
  const int hi   = lane >> 4;        // half-wave select
  const int c0   = hi * 8;

  v8f acc[2][4];
  v8f z = {};
  #pragma unroll
  for (int i = 0; i < 2; ++i)
    #pragma unroll
    for (int j = 0; j < 4; ++j) acc[i][j] = z;

  const unsigned short* A0 = A + (size_t)(m0 + l15) * K + c0;
  const unsigned short* A1 = A0 + (size_t)16 * K;
  const unsigned short* B0 = Bt + (size_t)(n0 + l15) * K + hi * 8;

  for (int k = 0; k < K; k += 32) {
    Frag a[2];
    a[0].q[0] = *(const uint4*)(A0 + k);
    a[0].q[1] = *(const uint4*)(A0 + k + 16);
    a[1].q[0] = *(const uint4*)(A1 + k);
    a[1].q[1] = *(const uint4*)(A1 + k + 16);
    #pragma unroll
    for (int in = 0; in < 4; ++in) {
      const unsigned short* Bp = B0 + (size_t)in * 16 * K + k;
      Frag bf;
      bf.q[0] = *(const uint4*)Bp;
      bf.q[1] = *(const uint4*)(Bp + 16);
      #pragma unroll
      for (int im = 0; im < 2; ++im)
        acc[im][in] = __builtin_amdgcn_wmma_f32_16x16x32_bf16(
            false, a[im].v, false, bf.v, (short)0, acc[im][in], false, false);
    }
  }

  #pragma unroll
  for (int im = 0; im < 2; ++im)
    #pragma unroll
    for (int in = 0; in < 4; ++in)
      #pragma unroll
      for (int r = 0; r < 8; ++r) {
        int row = m0 + im * 16 + r + hi * 8;
        int col = n0 + in * 16 + l15;
        float v = acc[im][in][r];
        if (OUTF32) ((float*)Cp)[(size_t)row * N + col] = v;
        else ((unsigned short*)Cp)[(size_t)row * N + col] = f2bf(v);
      }
}

// ---------------------------------------------------------------------------
// Repack fused QKV output [B*SD, 4096] into Q[b,h,s,d], K[b,kv,s,d], Vt[b,kv,d,s]
// ---------------------------------------------------------------------------
__global__ __launch_bounds__(256)
void repack_qkv_k(const unsigned short* __restrict__ qkv,
                  unsigned short* __restrict__ Qb,
                  unsigned short* __restrict__ Kb,
                  unsigned short* __restrict__ Vt) {
  size_t idx = (size_t)blockIdx.x * blockDim.x + threadIdx.x;  // B*SD*4096
  unsigned int c  = (unsigned int)(idx & 4095u);
  unsigned int sg = (unsigned int)(idx >> 12);
  unsigned int b  = sg >> 11, s = sg & 2047u;
  unsigned short v = qkv[idx];
  unsigned int d = c & 255u;
  if (c < 2048u) {
    unsigned int h = c >> 8;
    Qb[((size_t)(b * 8u + h) * 2048u + s) * 256u + d] = v;
  } else if (c < 3072u) {
    unsigned int kv = (c - 2048u) >> 8;
    Kb[((size_t)(b * 4u + kv) * 4096u + s) * 256u + d] = v;
  } else {
    unsigned int kv = (c - 3072u) >> 8;
    Vt[((size_t)(b * 4u + kv) * 256u + d) * 4096u + s] = v;
  }
}

// Cross K/V [B*SE, 1024] -> K rows SD.., Vt cols SD..
__global__ __launch_bounds__(256)
void repack_cross_k(const unsigned short* __restrict__ kc,
                    const unsigned short* __restrict__ vc,
                    unsigned short* __restrict__ Kb,
                    unsigned short* __restrict__ Vt) {
  size_t idx = (size_t)blockIdx.x * blockDim.x + threadIdx.x;  // B*SE*1024
  unsigned int c  = (unsigned int)(idx & 1023u);
  unsigned int sg = (unsigned int)(idx >> 10);
  unsigned int b  = sg >> 11, s = sg & 2047u;
  unsigned int kv = c >> 8, d = c & 255u;
  Kb[((size_t)(b * 4u + kv) * 4096u + 2048u + s) * 256u + d] = kc[idx];
  Vt[((size_t)(b * 4u + kv) * 256u + d) * 4096u + 2048u + s] = vc[idx];
}

// ---------------------------------------------------------------------------
// Flash attention: Q[b,h,2048,256] x K[b,kv,4096,256] -> softcap/mask/softmax
//                  -> x Vt[b,kv,256,4096] -> O[b,s,2048]  (bf16 in, f32 acc)
// 128 threads = 4 waves; each wave owns 16 q rows; key tiles of 32.
// ---------------------------------------------------------------------------
__global__ __launch_bounds__(128)
void flash_k(const unsigned short* __restrict__ Qb,
             const unsigned short* __restrict__ Kb,
             const unsigned short* __restrict__ Vt,
             unsigned short* __restrict__ Ob) {
  constexpr int SD = 2048, TOT = 4096, HD = 256, NH = 8, NKV = 4;
  const float SCALE = 0.0625f, CAPc = 50.0f, INVCAP = 0.02f, L2E = 1.44269504f;
  __shared__ __attribute__((aligned(16))) unsigned short lds_p[4][16][48];

  const int lane = threadIdx.x & 31;
  const int wid  = threadIdx.x >> 5;
  const int l15  = lane & 15;
  const int hi   = lane >> 4;

  int blk  = blockIdx.x;            // b*NH*32 + h*32 + qblk
  int qblk = blk & 31; blk >>= 5;
  int h    = blk & 7;
  int b    = blk >> 3;
  int kvh  = h >> 1;
  const int qbase = qblk * 64 + wid * 16;
  const int qhi   = qbase + 15;

  // Per-lane base pointers (Q row for the A-fragment, with half-wave k-offset folded in)
  const unsigned short* Qp =
      Qb + ((size_t)((b * NH + h) * 2048 + qbase + l15)) * HD + hi * 8;
  const unsigned short* Kbase = Kb + ((size_t)(b * NKV + kvh)) * TOT * HD;
  const unsigned short* Vbase = Vt + ((size_t)(b * NKV + kvh)) * HD * TOT;

  v8f o[16];
  v8f z = {};
  #pragma unroll
  for (int i = 0; i < 16; ++i) o[i] = z;
  float mrun[8], lrun[8];
  #pragma unroll
  for (int r = 0; r < 8; ++r) { mrun[r] = -1e30f; lrun[r] = 0.f; }

  for (int s0 = 0; s0 < TOT; s0 += 32) {
    if (s0 < SD && s0 > qhi) continue;   // fully-masked self tile

    // --- scores S = Q * K^T for a 16x32 tile (Q re-read from L2) ---
    v8f sa[2]; sa[0] = z; sa[1] = z;
    {
      const unsigned short* Kp0 = Kbase + ((size_t)(s0 + l15)) * HD + hi * 8;
      const unsigned short* Kp1 = Kp0 + (size_t)16 * HD;
      #pragma unroll
      for (int kk = 0; kk < 8; ++kk) {
        Frag qf, k0, k1;
        qf.q[0] = *(const uint4*)(Qp + kk * 32);
        qf.q[1] = *(const uint4*)(Qp + kk * 32 + 16);
        k0.q[0] = *(const uint4*)(Kp0 + kk * 32);
        k0.q[1] = *(const uint4*)(Kp0 + kk * 32 + 16);
        k1.q[0] = *(const uint4*)(Kp1 + kk * 32);
        k1.q[1] = *(const uint4*)(Kp1 + kk * 32 + 16);
        sa[0] = __builtin_amdgcn_wmma_f32_16x16x32_bf16(
            false, qf.v, false, k0.v, (short)0, sa[0], false, false);
        sa[1] = __builtin_amdgcn_wmma_f32_16x16x32_bf16(
            false, qf.v, false, k1.v, (short)0, sa[1], false, false);
      }
    }

    // --- scale, soft-cap, mask, online softmax (probs written back into sa) ---
    #pragma unroll
    for (int r = 0; r < 8; ++r) {
      int qrow = qbase + r + hi * 8;
      float mx = -1e30f;
      #pragma unroll
      for (int in = 0; in < 2; ++in) {
        float x = sa[in][r] * SCALE;
        x = CAPc * tanhf(x * INVCAP);
        if (s0 < SD) {
          int kidx = s0 + in * 16 + l15;
          if (kidx > qrow) x = -1e30f;
        }
        sa[in][r] = x;
        mx = fmaxf(mx, x);
      }
      #pragma unroll
      for (int d = 1; d < 16; d <<= 1) mx = fmaxf(mx, __shfl_xor(mx, d, 32));
      float mnew = fmaxf(mrun[r], mx);
      float f = exp2f((mrun[r] - mnew) * L2E);
      mrun[r] = mnew;
      float sum = 0.f;
      #pragma unroll
      for (int in = 0; in < 2; ++in) {
        float e = exp2f((sa[in][r] - mnew) * L2E);
        sa[in][r] = e;
        sum += e;
      }
      #pragma unroll
      for (int d = 1; d < 16; d <<= 1) sum += __shfl_xor(sum, d, 32);
      lrun[r] = lrun[r] * f + sum;
      #pragma unroll
      for (int dt = 0; dt < 16; ++dt) o[dt][r] *= f;
      #pragma unroll
      for (int in = 0; in < 2; ++in)
        lds_p[wid][r + hi * 8][in * 16 + l15] = f2bf(sa[in][r]);
    }

    // --- C-layout -> A-fragment shuffle through per-wave LDS ---
    asm volatile("s_wait_dscnt 0" ::: "memory");
    Frag pf;
    {
      const unsigned short* Lp = &lds_p[wid][l15][hi * 8];
      pf.q[0] = *(const uint4*)Lp;
      pf.q[1] = *(const uint4*)(Lp + 16);
    }

    // --- O += P * V, V fragments double-buffered for load/WMMA overlap ---
    {
      const unsigned short* Vp = Vbase + (size_t)l15 * TOT + s0 + hi * 8;
      Frag vb[2];
      vb[0].q[0] = *(const uint4*)Vp;
      vb[0].q[1] = *(const uint4*)(Vp + 16);
      #pragma unroll
      for (int dt = 0; dt < 16; ++dt) {
        if (dt < 15) {
          const unsigned short* Vn = Vp + (size_t)(dt + 1) * 16 * TOT;
          vb[(dt + 1) & 1].q[0] = *(const uint4*)Vn;
          vb[(dt + 1) & 1].q[1] = *(const uint4*)(Vn + 16);
        }
        o[dt] = __builtin_amdgcn_wmma_f32_16x16x32_bf16(
            false, pf.v, false, vb[dt & 1].v, (short)0, o[dt], false, false);
      }
    }
  }

  // --- epilogue: O /= l, store bf16 to [B,SD,NH*HD] ---
  float inv[8];
  #pragma unroll
  for (int r = 0; r < 8; ++r) inv[r] = 1.0f / lrun[r];
  #pragma unroll
  for (int dt = 0; dt < 16; ++dt)
    #pragma unroll
    for (int r = 0; r < 8; ++r) {
      int qrow = qbase + r + hi * 8;
      Ob[((size_t)(b * SD + qrow)) * 2048 + h * 256 + dt * 16 + l15] =
          f2bf(o[dt][r] * inv[r]);
    }
}

// ---------------------------------------------------------------------------
// Host launch
// ---------------------------------------------------------------------------
extern "C" void kernel_launch(void* const* d_in, const int* in_sizes, int n_in,
                              void* d_out, int out_size, void* d_ws, size_t ws_size,
                              hipStream_t stream) {
  (void)in_sizes; (void)n_in; (void)out_size; (void)ws_size;
  const float* hidden = (const float*)d_in[0];   // [2,2048,2304]
  const float* enc    = (const float*)d_in[1];   // [2,2048,2304]
  const float* w_qkv  = (const float*)d_in[3];   // [2304,4096]
  const float* w_k    = (const float*)d_in[4];   // [2304,1024]
  const float* w_v    = (const float*)d_in[5];   // [2304,1024]
  const float* w_o    = (const float*)d_in[6];   // [2048,2304]
  float* out = (float*)d_out;                    // [2,2048,2304]

  unsigned short* ws = (unsigned short*)d_ws;    // element offsets (2B each)
  const size_t o_wqkvt = 0;                                // 4096*2304
  const size_t o_wkt   = o_wqkvt + (size_t)4096 * 2304;    // 1024*2304
  const size_t o_wvt   = o_wkt   + (size_t)1024 * 2304;    // 1024*2304
  const size_t o_wot   = o_wvt   + (size_t)1024 * 2304;    // 2304*2048
  const size_t o_hsb   = o_wot   + (size_t)2304 * 2048;    // 4096*2304
  const size_t o_encb  = o_hsb   + (size_t)4096 * 2304;    // 4096*2304
  const size_t o_qkv   = o_encb  + (size_t)4096 * 2304;    // 4096*4096
  const size_t o_kc    = o_qkv   + (size_t)4096 * 4096;    // 4096*1024
  const size_t o_vc    = o_kc    + (size_t)4096 * 1024;    // 4096*1024
  const size_t o_qb    = o_vc    + (size_t)4096 * 1024;    // 8388608
  const size_t o_kb    = o_qb    + (size_t)8388608;        // 8388608
  const size_t o_vt    = o_kb    + (size_t)8388608;        // 8388608
  unsigned short* wqkvt = ws + o_wqkvt;
  unsigned short* wkt   = ws + o_wkt;
  unsigned short* wvt   = ws + o_wvt;
  unsigned short* wot   = ws + o_wot;
  unsigned short* hsb   = ws + o_hsb;
  unsigned short* encb  = ws + o_encb;
  unsigned short* qkv   = ws + o_qkv;
  unsigned short* kc    = ws + o_kc;
  unsigned short* vc    = ws + o_vc;
  unsigned short* qb    = ws + o_qb;
  unsigned short* kb    = ws + o_kb;
  unsigned short* vt    = ws + o_vt;
  unsigned short* attn  = ws + o_qkv;  // alias dead qkv buffer (needs 4096*2048)

  // 1) activation + weight conversions (weights transposed to [N,K])
  ccvt_k<<<(4096u * 2304u) / 256u, 256, 0, stream>>>(hidden, hsb, (size_t)4096 * 2304);
  ccvt_k<<<(4096u * 2304u) / 256u, 256, 0, stream>>>(enc,    encb, (size_t)4096 * 2304);
  tcvt_k<<<(2304u * 4096u) / 256u, 256, 0, stream>>>(w_qkv, wqkvt, 2304, 4096);
  tcvt_k<<<(2304u * 1024u) / 256u, 256, 0, stream>>>(w_k,   wkt,   2304, 1024);
  tcvt_k<<<(2304u * 1024u) / 256u, 256, 0, stream>>>(w_v,   wvt,   2304, 1024);
  tcvt_k<<<(2048u * 2304u) / 256u, 256, 0, stream>>>(w_o,   wot,   2048, 2304);

  // 2) projections (pure bf16 WMMA GEMMs)
  gemm_k<0><<<dim3(32, 32), 256, 0, stream>>>(hsb,  wqkvt, qkv, 4096, 4096, 2304);
  gemm_k<0><<<dim3(8, 32),  256, 0, stream>>>(encb, wkt,   kc,  4096, 1024, 2304);
  gemm_k<0><<<dim3(8, 32),  256, 0, stream>>>(encb, wvt,   vc,  4096, 1024, 2304);

  // 3) repack into attention layouts (V stored transposed)
  repack_qkv_k<<<(4096u * 4096u) / 256u, 256, 0, stream>>>(qkv, qb, kb, vt);
  repack_cross_k<<<(4096u * 1024u) / 256u, 256, 0, stream>>>(kc, vc, kb, vt);

  // 4) flash attention
  flash_k<<<512, 128, 0, stream>>>(qb, kb, vt, attn);

  // 5) output projection (f32 epilogue -> d_out)
  gemm_k<1><<<dim3(18, 32), 256, 0, stream>>>(attn, wot, out, 4096, 2304, 2048);
}
